// FluxAttention_17772574670979
// MI455X (gfx1250) — matrix operand order repaired and checked by reading, not compiled, not run on hardware
//
#include <hip/hip_runtime.h>
#include <hip/hip_bf16.h>

// ---------------------------------------------------------------------------
// MSDeformAttention (separable offsets) for MI455X / gfx1250, wave32 + WMMA.
// Pipeline: pack_weights -> value_proj (WMMA, M=32) -> fused offsets+sampling
// (WMMA 288-col query projection, lane=DH bilinear gather) -> out_proj (WMMA).
// GEMM loops are k-outer / tile-inner: accumulators (4x v8f) stay resident,
// A fragments are rebuilt from LDS per k-step (each feeds 2 WMMAs), and each
// B fragment feeds 2 WMMAs -> no scratch spills, halved weight traffic.
// ---------------------------------------------------------------------------

typedef __attribute__((ext_vector_type(16))) __bf16 v16bf;
typedef __attribute__((ext_vector_type(8)))  float  v8f;

#define B_    4
#define NQ_   8192
#define D_    256
#define H_    8
#define DH_   32
#define L_    4
#define NV_   5440
#define NT_CAT 18           // 288/16 column tiles
#define NT_D   16           // 256/16 column tiles
#define KSTEPS 8            // 256/32

// workspace byte offsets (all 256-aligned)
#define WS_WCAT   ((size_t)0)                       // 18*8*32*16 bf16 = 147456 B
#define WS_WV     ((size_t)147456)                  // 16*8*32*16 bf16 = 131072 B
#define WS_WO     ((size_t)278528)                  // 131072 B
#define WS_BCAT   ((size_t)409600)                  // 288 f32
#define WS_VPROJ  ((size_t)410880)                  // B*H*NV*DH f32 = 22282240 B
#define WS_OHEAD  ((size_t)22693120)                // B*NQ*H*DH f32 = 33554432 B

__device__ __forceinline__ unsigned short f2bf_bits(float f) {
  unsigned u = __builtin_bit_cast(unsigned, f);
  u = (u + 0x7FFFu + ((u >> 16) & 1u)) >> 16;       // RNE
  return (unsigned short)u;
}
__device__ __forceinline__ __bf16 f2bf(float f) {
  unsigned short s = f2bf_bits(f);
  return __builtin_bit_cast(__bf16, s);
}

// A fragment (16x32 bf16) from an LDS tile row block, contiguous k order (same as the B pack):
// lanes 0-15 (row=lane):     elems 0-15 -> K 0..15
// lanes 16-31 (row=lane-16): elems 0-15 -> K 16..31
__device__ __forceinline__ v16bf make_a_frag(const __bf16 (*qa)[264], int rowoff,
                                             int kbase, int lane) {
  const int m  = rowoff + (lane & 15);
  const int hi = (lane >> 4) & 1;
  const __bf16* row = &qa[m][kbase + hi * 16];
  v16bf f;
#pragma unroll
  for (int e = 0; e < 8; ++e) f[e] = row[e];
#pragma unroll
  for (int e = 0; e < 8; ++e) f[8 + e] = row[8 + e];
  return f;
}

// ---------------------------------------------------------------------------
// Kernel 1: pack weights to bf16 B-fragment layout.
// dest index = ((tile*8 + kstep)*32 + lane)*16 + e
//   column = tile*16 + (lane&15);  K = kstep*32 + (lane>=16)*16 + e
// ---------------------------------------------------------------------------
__global__ void pack_weights_kernel(const float* __restrict__ Wt, const float* __restrict__ Wf,
                                    const float* __restrict__ Wl, const float* __restrict__ Wp,
                                    const float* __restrict__ Wv, const float* __restrict__ Wo,
                                    const float* __restrict__ bt, const float* __restrict__ bfr,
                                    const float* __restrict__ bl, const float* __restrict__ bp,
                                    unsigned short* __restrict__ wcat,
                                    unsigned short* __restrict__ wv,
                                    unsigned short* __restrict__ wo,
                                    float* __restrict__ bcat) {
  const int tid = blockIdx.x * 256 + threadIdx.x;
  const int NCAT = NT_CAT * KSTEPS * 32 * 16;       // 73728
  const int N256 = NT_D * KSTEPS * 32 * 16;         // 65536

  if (tid < 288) {
    float v;
    if      (tid < 64)  v = bt[tid];
    else if (tid < 128) v = bfr[tid - 64];
    else if (tid < 160) v = bl[tid - 128];
    else                v = bp[tid - 160];
    bcat[tid] = v;
  }

  if (tid < NCAT) {
    const int e  = tid & 15;
    const int ln = (tid >> 4) & 31;
    const int ks = (tid >> 9) & 7;
    const int t  = tid >> 12;
    const int col = t * 16 + (ln & 15);
    const int k   = ks * 32 + ((ln >> 4) & 1) * 16 + e;
    float v;
    if      (col < 64)  v = Wt[k * 64 + col];
    else if (col < 128) v = Wf[k * 64 + (col - 64)];
    else if (col < 160) v = Wl[k * 32 + (col - 128)];
    else                v = Wp[k * 128 + (col - 160)];
    wcat[tid] = f2bf_bits(v);
  } else if (tid < NCAT + 2 * N256) {
    const int id  = tid - NCAT;
    const int sel = id / N256;                      // 0 -> Wv, 1 -> Wo
    const int loc = id % N256;
    const int e  = loc & 15;
    const int ln = (loc >> 4) & 31;
    const int ks = (loc >> 9) & 7;
    const int t  = loc >> 12;
    const int col = t * 16 + (ln & 15);
    const int k   = ks * 32 + ((ln >> 4) & 1) * 16 + e;
    if (sel == 0) wv[loc] = f2bf_bits(Wv[k * D_ + col]);
    else          wo[loc] = f2bf_bits(Wo[k * D_ + col]);
  }
}

// k-outer GEMM over two column tiles t0,t1 and two 16-row halves.
// accs: [t0/rows0-15, t0/rows16-31, t1/rows0-15, t1/rows16-31]
__device__ __forceinline__ void gemm_2tiles(const __bf16 (*qa)[264],
                                            const v16bf* __restrict__ bfrags,
                                            int t0, int t1, int lane,
                                            v8f& a00, v8f& a01, v8f& a10, v8f& a11) {
#pragma unroll
  for (int ks = 0; ks < KSTEPS; ++ks) {
    if (ks < KSTEPS - 1) {
      __builtin_prefetch(&bfrags[(t0 * KSTEPS + ks + 1) * 32 + lane], 0, 3);
      __builtin_prefetch(&bfrags[(t1 * KSTEPS + ks + 1) * 32 + lane], 0, 3);
    }
    v16bf fa0 = make_a_frag(qa, 0,  ks * 32, lane);
    v16bf fa1 = make_a_frag(qa, 16, ks * 32, lane);
    v16bf b0  = bfrags[(t0 * KSTEPS + ks) * 32 + lane];
    v16bf b1  = bfrags[(t1 * KSTEPS + ks) * 32 + lane];
    a00 = __builtin_amdgcn_wmma_f32_16x16x32_bf16(false, fa0, false, b0, (short)0, a00, false, false);
    a01 = __builtin_amdgcn_wmma_f32_16x16x32_bf16(false, fa1, false, b0, (short)0, a01, false, false);
    a10 = __builtin_amdgcn_wmma_f32_16x16x32_bf16(false, fa0, false, b1, (short)0, a10, false, false);
    a11 = __builtin_amdgcn_wmma_f32_16x16x32_bf16(false, fa1, false, b1, (short)0, a11, false, false);
  }
}

__device__ __forceinline__ void gemm_1tile(const __bf16 (*qa)[264],
                                           const v16bf* __restrict__ bfrags,
                                           int t, int lane, v8f& a0, v8f& a1) {
#pragma unroll
  for (int ks = 0; ks < KSTEPS; ++ks) {
    if (ks < KSTEPS - 1) __builtin_prefetch(&bfrags[(t * KSTEPS + ks + 1) * 32 + lane], 0, 3);
    v16bf fa0 = make_a_frag(qa, 0,  ks * 32, lane);
    v16bf fa1 = make_a_frag(qa, 16, ks * 32, lane);
    v16bf bb  = bfrags[(t * KSTEPS + ks) * 32 + lane];
    a0 = __builtin_amdgcn_wmma_f32_16x16x32_bf16(false, fa0, false, bb, (short)0, a0, false, false);
    a1 = __builtin_amdgcn_wmma_f32_16x16x32_bf16(false, fa1, false, bb, (short)0, a1, false, false);
  }
}

// ---------------------------------------------------------------------------
// Kernel 2: v = value @ W_v + b_v, stored head-major as [B][H][NV][DH] f32.
// grid (NV/32, B), 256 threads = 8 waves; each wave owns 2 column tiles.
// ---------------------------------------------------------------------------
__global__ void value_proj_kernel(const float* __restrict__ value,
                                  const unsigned short* __restrict__ wvpack,
                                  const float* __restrict__ bv,
                                  float* __restrict__ vproj) {
  __shared__ __bf16 qa[32][264];
  const int b  = blockIdx.y;
  const int n0 = blockIdx.x * 32;
  const int tid = threadIdx.x;

#pragma unroll
  for (int j = 0; j < 32; ++j) {
    const int idx = tid + 256 * j;
    const int row = idx >> 8, col = idx & 255;
    qa[row][col] = f2bf(value[((size_t)b * NV_ + (n0 + row)) * D_ + col]);
  }
  __syncthreads();

  const int w = tid >> 5, lane = tid & 31;
  const v16bf* bfrags = reinterpret_cast<const v16bf*>(wvpack);

  const int t0 = w * 2, t1 = w * 2 + 1;
  v8f a00 = {}, a01 = {}, a10 = {}, a11 = {};
  gemm_2tiles(qa, bfrags, t0, t1, lane, a00, a01, a10, a11);

  const int n = lane & 15;
  const int mbase = (lane >> 4) << 3;
#pragma unroll
  for (int half = 0; half < 2; ++half) {
    const int t = half ? t1 : t0;
    const int c = t * 16 + n;
    const int h = c >> 5, dh = c & 31;
    const float bias = bv[c];
    const v8f& lo = half ? a10 : a00;
    const v8f& hi = half ? a11 : a01;
#pragma unroll
    for (int r = 0; r < 8; ++r) {
      const int m = mbase + r;
      vproj[(((size_t)b * H_ + h) * NV_ + (n0 + m)) * DH_ + dh]      = lo[r] + bias;
      vproj[(((size_t)b * H_ + h) * NV_ + (n0 + 16 + m)) * DH_ + dh] = hi[r] + bias;
    }
  }
}

// ---------------------------------------------------------------------------
// Kernel 3: fused query projection (WMMA, 288 cols, M=32) + tanh/softmax +
// bilinear sampling. grid (NQ/32, B), 256 threads = 8 waves.
// Tiles 0-15: 2 per wave; tiles 16-17: second pass on waves 0-1.
// Phase B: wave handles 32 (q,h) pairs; lane = dh (DH == wave32).
// ---------------------------------------------------------------------------
__global__ void msda_fused_kernel(const float* __restrict__ query,
                                  const float* __restrict__ refp,
                                  const unsigned short* __restrict__ wcatpack,
                                  const float* __restrict__ bcat,
                                  const float* __restrict__ vproj,
                                  float* __restrict__ out_heads) {
  __shared__ __bf16 qa[32][264];
  __shared__ float  S[32][292];
  const int b  = blockIdx.y;
  const int q0 = blockIdx.x * 32;
  const int tid = threadIdx.x;

#pragma unroll
  for (int j = 0; j < 32; ++j) {
    const int idx = tid + 256 * j;
    const int row = idx >> 8, col = idx & 255;
    qa[row][col] = f2bf(query[((size_t)b * NQ_ + (q0 + row)) * D_ + col]);
  }
  __syncthreads();

  const int w = tid >> 5, lane = tid & 31;
  const v16bf* bfrags = reinterpret_cast<const v16bf*>(wcatpack);
  const int n = lane & 15;
  const int mbase = (lane >> 4) << 3;

  {
    const int t0 = w * 2, t1 = w * 2 + 1;
    v8f a00 = {}, a01 = {}, a10 = {}, a11 = {};
    gemm_2tiles(qa, bfrags, t0, t1, lane, a00, a01, a10, a11);
#pragma unroll
    for (int half = 0; half < 2; ++half) {
      const int t = half ? t1 : t0;
      const int c = t * 16 + n;
      const float bias = bcat[c];
      const v8f& lo = half ? a10 : a00;
      const v8f& hi = half ? a11 : a01;
#pragma unroll
      for (int r = 0; r < 8; ++r) {
        S[mbase + r][c]      = lo[r] + bias;
        S[16 + mbase + r][c] = hi[r] + bias;
      }
    }
  }
  if (w < 2) {                           // tiles 16,17 (wave-uniform branch)
    const int t = 16 + w;
    v8f e0 = {}, e1 = {};
    gemm_1tile(qa, bfrags, t, lane, e0, e1);
    const int c = t * 16 + n;
    const float bias = bcat[c];
#pragma unroll
    for (int r = 0; r < 8; ++r) {
      S[mbase + r][c]      = e0[r] + bias;
      S[16 + mbase + r][c] = e1[r] + bias;
    }
  }
  __syncthreads();

  const int Hls[L_] = {64, 32, 16, 8};
  const int Wls[L_] = {64, 32, 16, 8};
  const int lsi[L_] = {0, 4096, 5120, 5376};

  for (int i = 0; i < 32; ++i) {
    const int p  = w * 32 + i;          // 256 pairs = 32 queries x 8 heads
    const int ql = p >> 3;
    const int h  = p & 7;
    const int q  = q0 + ql;
    const float* Sq = &S[ql][0];

    // level softmax
    float ll[L_];
    float lmax = -1e30f;
#pragma unroll
    for (int l = 0; l < L_; ++l) { ll[l] = Sq[128 + h * L_ + l]; lmax = fmaxf(lmax, ll[l]); }
    float lsum = 0.f;
#pragma unroll
    for (int l = 0; l < L_; ++l) { ll[l] = __expf(ll[l] - lmax); lsum += ll[l]; }
    const float invls = 1.f / lsum;

    float acc = 0.f;
#pragma unroll
    for (int l = 0; l < L_; ++l) {
      const int base2 = (h * L_ + l) * 2;
      const float tx[2] = {tanhf(Sq[base2 + 0]),      tanhf(Sq[base2 + 1])};
      const float ty[2] = {tanhf(Sq[64 + base2 + 0]), tanhf(Sq[64 + base2 + 1])};
      // point softmax
      float pl[4];
      float pmax = -1e30f;
#pragma unroll
      for (int k = 0; k < 4; ++k) { pl[k] = Sq[160 + (h * L_ + l) * 4 + k]; pmax = fmaxf(pmax, pl[k]); }
      float psum = 0.f;
#pragma unroll
      for (int k = 0; k < 4; ++k) { pl[k] = __expf(pl[k] - pmax); psum += pl[k]; }
      const float invps = invls / psum;   // lw*pw = ll[l]*pl[k]*invls/psum

      const float rx = refp[(((size_t)b * NQ_ + q) * L_ + l) * 2 + 0];
      const float ry = refp[(((size_t)b * NQ_ + q) * L_ + l) * 2 + 1];
      const int   Wi = Wls[l], Hi = Hls[l];
      const float Wfv = (float)Wi, Hfv = (float)Hi;
      const float* vlev = vproj + (((size_t)b * H_ + h) * NV_ + lsi[l]) * DH_;
      const float lscale = ll[l] * invps;

#pragma unroll
      for (int kt = 0; kt < 2; ++kt) {
#pragma unroll
        for (int kf = 0; kf < 2; ++kf) {
          const int   k = kt * 2 + kf;
          const float a = lscale * pl[k];
          float lx = rx + tx[kt] / Wfv; lx = fminf(fmaxf(lx, 0.f), 1.f);
          float ly = ry + ty[kf] / Hfv; ly = fminf(fmaxf(ly, 0.f), 1.f);
          const float px = lx * Wfv - 0.5f;
          const float py = ly * Hfv - 0.5f;
          const float fx0 = floorf(px), fy0 = floorf(py);
          const float wx = px - fx0, wy = py - fy0;
          const int ix = (int)fx0, iy = (int)fy0;

          auto samp = [&](int X, int Y, float wgt) {
            if (X >= 0 && X < Wi && Y >= 0 && Y < Hi)
              acc += wgt * vlev[((size_t)(Y * Wi + X)) * DH_ + lane];
          };
          samp(ix,     iy,     (1.f - wx) * (1.f - wy) * a);
          samp(ix + 1, iy,     wx         * (1.f - wy) * a);
          samp(ix,     iy + 1, (1.f - wx) * wy         * a);
          samp(ix + 1, iy + 1, wx         * wy         * a);
        }
      }
    }
    out_heads[(((size_t)b * NQ_ + q) * H_ + h) * DH_ + lane] = acc;
  }
}

// ---------------------------------------------------------------------------
// Kernel 4: out = out_heads @ W_o + b_o.  grid (NQ/32, B), M=32.
// ---------------------------------------------------------------------------
__global__ void out_proj_kernel(const float* __restrict__ out_heads,
                                const unsigned short* __restrict__ wopack,
                                const float* __restrict__ bo,
                                float* __restrict__ out) {
  __shared__ __bf16 qa[32][264];
  const int b  = blockIdx.y;
  const int q0 = blockIdx.x * 32;
  const int tid = threadIdx.x;

#pragma unroll
  for (int j = 0; j < 32; ++j) {
    const int idx = tid + 256 * j;
    const int row = idx >> 8, col = idx & 255;
    qa[row][col] = f2bf(out_heads[((size_t)b * NQ_ + (q0 + row)) * D_ + col]);
  }
  __syncthreads();

  const int w = tid >> 5, lane = tid & 31;
  const v16bf* bfrags = reinterpret_cast<const v16bf*>(wopack);

  const int t0 = w * 2, t1 = w * 2 + 1;
  v8f a00 = {}, a01 = {}, a10 = {}, a11 = {};
  gemm_2tiles(qa, bfrags, t0, t1, lane, a00, a01, a10, a11);

  const int n = lane & 15;
  const int mbase = (lane >> 4) << 3;
#pragma unroll
  for (int half = 0; half < 2; ++half) {
    const int t = half ? t1 : t0;
    const int c = t * 16 + n;
    const float bias = bo[c];
    const v8f& lo = half ? a10 : a00;
    const v8f& hi = half ? a11 : a01;
#pragma unroll
    for (int r = 0; r < 8; ++r) {
      const int m = mbase + r;
      out[((size_t)b * NQ_ + (q0 + m)) * D_ + c]      = lo[r] + bias;
      out[((size_t)b * NQ_ + (q0 + 16 + m)) * D_ + c] = hi[r] + bias;
    }
  }
}

// ---------------------------------------------------------------------------
extern "C" void kernel_launch(void* const* d_in, const int* in_sizes, int n_in,
                              void* d_out, int out_size, void* d_ws, size_t ws_size,
                              hipStream_t stream) {
  (void)in_sizes; (void)n_in; (void)out_size; (void)ws_size;
  const float* query = (const float*)d_in[0];
  const float* refp  = (const float*)d_in[1];
  const float* value = (const float*)d_in[2];
  // d_in[3] spatial_shapes, d_in[4] level_start_index: compile-time constants here
  const float* W_time = (const float*)d_in[5];
  const float* b_time = (const float*)d_in[6];
  const float* W_freq = (const float*)d_in[7];
  const float* b_freq = (const float*)d_in[8];
  const float* W_lvl  = (const float*)d_in[9];
  const float* b_lvl  = (const float*)d_in[10];
  const float* W_pt   = (const float*)d_in[11];
  const float* b_pt   = (const float*)d_in[12];
  const float* W_v    = (const float*)d_in[13];
  const float* b_v    = (const float*)d_in[14];
  const float* W_o    = (const float*)d_in[15];
  const float* b_o    = (const float*)d_in[16];

  char* ws = (char*)d_ws;
  unsigned short* wcat = (unsigned short*)(ws + WS_WCAT);
  unsigned short* wv   = (unsigned short*)(ws + WS_WV);
  unsigned short* wo   = (unsigned short*)(ws + WS_WO);
  float* bcat          = (float*)(ws + WS_BCAT);
  float* vproj         = (float*)(ws + WS_VPROJ);
  float* oheads        = (float*)(ws + WS_OHEAD);

  pack_weights_kernel<<<800, 256, 0, stream>>>(W_time, W_freq, W_lvl, W_pt, W_v, W_o,
                                               b_time, b_freq, b_lvl, b_pt,
                                               wcat, wv, wo, bcat);
  value_proj_kernel<<<dim3(NV_ / 32, B_), 256, 0, stream>>>(value, wv, b_v, vproj);
  msda_fused_kernel<<<dim3(NQ_ / 32, B_), 256, 0, stream>>>(query, refp, wcat, bcat,
                                                            vproj, oheads);
  out_proj_kernel<<<dim3(NQ_ / 32, B_), 256, 0, stream>>>(oheads, wo, b_o, (float*)d_out);
}
